// SA_28200755265724
// MI455X (gfx1250) — matrix operand order repaired and checked by reading, not compiled, not run on hardware
//
#include <hip/hip_runtime.h>
#include <hip/hip_bf16.h>

// ---------------------------------------------------------------------------
// PointNet++ SA layer for MI455X (gfx1250, wave32, WMMA)
//   N=8192 points, K=128 centers, D=3, IFD=64, L1=64, L2=64, L3=128, r=1
// Pipeline:
//   k1: FPS (serial argmax loop, 1 block)             -> centers
//   k2: pack W1feat/W2/W3 into WMMA B-fragment layout -> f16 in ws
//   k3: P = features @ W1[3:,:] + b1  (WMMA f16)      -> f32 in ws
//   k4: fused per-center MLP2/MLP3 + masked max (WMMA f16, LDS-staged)
// ---------------------------------------------------------------------------

typedef _Float16 v16h __attribute__((ext_vector_type(16)));
typedef _Float16 v8h  __attribute__((ext_vector_type(8)));
typedef float    v8f  __attribute__((ext_vector_type(8)));

#define NPTS   8192
#define KCEN   128
#define L3DIM  128
#define H1STR  72          // padded f16 row stride: 16B-aligned rows, bank-conflict-free
#define NEG_INF (-__builtin_inff())

// A fragment (16x32 f16) from LDS: two contiguous 8-half (16B) chunks per lane.
__device__ __forceinline__ v16h afrag_lds(const _Float16* p0, const _Float16* p1) {
  v8h a0 = *(const v8h*)p0;
  v8h a1 = *(const v8h*)p1;
  v16h r;
#pragma unroll
  for (int e = 0; e < 8; ++e) { r[e] = a0[e]; r[e + 8] = a1[e]; }
  return r;
}

// A fragment from row-major f32 global memory with on-the-fly f16 convert.
__device__ __forceinline__ v16h afrag_f32(const float* p0, const float* p1) {
  v16h r;
#pragma unroll
  for (int e = 0; e < 8; ++e) { r[e] = (_Float16)p0[e]; r[e + 8] = (_Float16)p1[e]; }
  return r;
}

__device__ __forceinline__ v8f wmma16(v16h a, v16h b, v8f c) {
  return __builtin_amdgcn_wmma_f32_16x16x32_f16(false, a, false, b, (short)0, c,
                                                false, false);
}

// ---------------------------------------------------------------------------
// Kernel 1: farthest point sampling (1 block x 1024 threads, 8 pts/thread)
// ---------------------------------------------------------------------------
__global__ __launch_bounds__(1024) void fps_kernel(const float* __restrict__ coords,
                                                   float* __restrict__ centersWs,
                                                   float* __restrict__ centersOut) {
  __shared__ float sval[32];
  __shared__ int   sidx[32];
  __shared__ float scx, scy, scz;
  __shared__ int   slast;
  const int tid = threadIdx.x, lane = tid & 31, wv = tid >> 5;

  float px[8], py[8], pz[8], dist[8];
#pragma unroll
  for (int j = 0; j < 8; ++j) {
    int p = tid + j * 1024;
    px[j] = coords[p * 3 + 0];
    py[j] = coords[p * 3 + 1];
    pz[j] = coords[p * 3 + 2];
    dist[j] = 1e10f;
  }
  if (tid == 0) {
    slast = 0;
    centersWs[0] = coords[0];  centersWs[1] = coords[1];  centersWs[2] = coords[2];
    centersOut[0] = coords[0]; centersOut[1] = coords[1]; centersOut[2] = coords[2];
  }
  __syncthreads();

  for (int i = 1; i < KCEN; ++i) {
    if (tid == 0) {
      int l = slast;
      scx = coords[l * 3 + 0]; scy = coords[l * 3 + 1]; scz = coords[l * 3 + 2];
    }
    __syncthreads();
    const float cx = scx, cy = scy, cz = scz;

    float bv = -1.0f; int bi = 0;
#pragma unroll
    for (int j = 0; j < 8; ++j) {
      float dx = px[j] - cx, dy = py[j] - cy, dz = pz[j] - cz;
      float d2 = dx * dx + dy * dy + dz * dz;
      float nd = fminf(dist[j], d2);
      dist[j] = nd;
      if (nd > bv) { bv = nd; bi = tid + j * 1024; }  // ascending j => first-max kept
    }
    // wave32 reduce, tie-break lowest index (match jnp.argmax)
#pragma unroll
    for (int off = 16; off > 0; off >>= 1) {
      float ov = __shfl_down(bv, off, 32);
      int   oi = __shfl_down(bi, off, 32);
      if (ov > bv || (ov == bv && oi < bi)) { bv = ov; bi = oi; }
    }
    if (lane == 0) { sval[wv] = bv; sidx[wv] = bi; }
    __syncthreads();
    if (wv == 0) {
      float v = sval[lane]; int ix = sidx[lane];
#pragma unroll
      for (int off = 16; off > 0; off >>= 1) {
        float ov = __shfl_down(v, off, 32);
        int   oi = __shfl_down(ix, off, 32);
        if (ov > v || (ov == v && oi < ix)) { v = ov; ix = oi; }
      }
      if (lane == 0) {
        slast = ix;
        float x = coords[ix * 3], y = coords[ix * 3 + 1], z = coords[ix * 3 + 2];
        centersWs[i * 3 + 0] = x; centersWs[i * 3 + 1] = y; centersWs[i * 3 + 2] = z;
        centersOut[i * 3 + 0] = x; centersOut[i * 3 + 1] = y; centersOut[i * 3 + 2] = z;
      }
    }
    __syncthreads();
  }
}

// ---------------------------------------------------------------------------
// Kernel 2: pack W1[3:,:], W2, W3 into WMMA B-fragment layout (f16).
// B 32x16 tile layout: lane l holds column n = l%16, 16 K values (table order)
// K = kt*32 + 8*(l/16) + (e%8) + 16*(e/8)  -> packed[( (kt*NT+nt)*32 + l )*16 + e]
// ---------------------------------------------------------------------------
__global__ __launch_bounds__(256) void pack_weights(const float* __restrict__ W1,
                                                    const float* __restrict__ W2,
                                                    const float* __restrict__ W3,
                                                    _Float16* __restrict__ pw1,
                                                    _Float16* __restrict__ pw2,
                                                    _Float16* __restrict__ pw3) {
  int t = blockIdx.x * 256 + threadIdx.x;
  if (t < 4096) {                       // pw1 & pw2: K=64 (2 ktiles), N=64 (4 ntiles)
    int e = t & 15, l = (t >> 4) & 31, nt = (t >> 9) & 3, kt = (t >> 11) & 1;
    int k = kt * 32 + ((l >> 4) << 3) + (e & 7) + ((e >> 3) << 4);
    int n = nt * 16 + (l & 15);
    pw1[t] = (_Float16)W1[(3 + k) * 64 + n];
    pw2[t] = (_Float16)W2[k * 64 + n];
  } else if (t < 4096 + 8192) {         // pw3: K=64 (2 ktiles), N=128 (8 ntiles)
    int u = t - 4096;
    int e = u & 15, l = (u >> 4) & 31, nt = (u >> 9) & 7, kt = (u >> 12) & 1;
    int k = kt * 32 + ((l >> 4) << 3) + (e & 7) + ((e >> 3) << 4);
    int n = nt * 16 + (l & 15);
    pw3[u] = (_Float16)W3[k * 128 + n];
  }
}

// ---------------------------------------------------------------------------
// Kernel 3: P[8192,64] = features @ W1[3:,:] + b1   (no ReLU; f32 out)
// 64 blocks x 256 threads (8 waves); each wave: 16 rows x 64 cols, 8 WMMAs.
// ---------------------------------------------------------------------------
__global__ __launch_bounds__(256) void mlp1_feat(const float* __restrict__ features,
                                                 const float* __restrict__ b1,
                                                 const _Float16* __restrict__ pw1,
                                                 float* __restrict__ P) {
  const int lane = threadIdx.x & 31, wv = threadIdx.x >> 5;
  const int half = lane >> 4, col = lane & 15;
  const int row0 = blockIdx.x * 128 + wv * 16;

  v16h w1f[2][4];
#pragma unroll
  for (int kt = 0; kt < 2; ++kt)
#pragma unroll
    for (int nt = 0; nt < 4; ++nt)
      w1f[kt][nt] = *(const v16h*)(pw1 + ((kt * 4 + nt) * 32 + lane) * 16);

  v8f acc[4];
#pragma unroll
  for (int nt = 0; nt < 4; ++nt) {
    float b = b1[nt * 16 + col];
#pragma unroll
    for (int v = 0; v < 8; ++v) acc[nt][v] = b;
  }

  const float* rowp = features + (size_t)(row0 + col) * 64;
#pragma unroll
  for (int kt = 0; kt < 2; ++kt) {
    int c0 = kt * 32 + 8 * half;
    v16h a = afrag_f32(rowp + c0, rowp + c0 + 16);
#pragma unroll
    for (int nt = 0; nt < 4; ++nt) acc[nt] = wmma16(a, w1f[kt][nt], acc[nt]);
  }

#pragma unroll
  for (int nt = 0; nt < 4; ++nt)
#pragma unroll
    for (int v = 0; v < 8; ++v)
      P[(size_t)(row0 + v + 8 * half) * 64 + nt * 16 + col] = acc[nt][v];
}

// ---------------------------------------------------------------------------
// Kernel 4: fused per-center MLP2+MLP3 + masked max.
// 128 blocks (1 per center) x 256 threads (8 waves), 128-point tiles.
// ---------------------------------------------------------------------------
__global__ __launch_bounds__(256) void sa_fused(const float* __restrict__ coords,
                                                const float* __restrict__ P,
                                                const float* __restrict__ W1,
                                                const float* __restrict__ b2v,
                                                const float* __restrict__ b3v,
                                                const _Float16* __restrict__ pw2,
                                                const _Float16* __restrict__ pw3,
                                                const float* __restrict__ centers,
                                                float* __restrict__ outFeat) {
  __shared__ __align__(16) _Float16 h1ls[128 * H1STR];
  __shared__ __align__(16) _Float16 h2ls[128 * H1STR];
  __shared__ float maskAdd[128];
  __shared__ float w1c[3 * 64];
  __shared__ float red[8][128];

  const int tid = threadIdx.x, lane = tid & 31, wv = tid >> 5;
  const int half = lane >> 4, col = lane & 15;
  const int kc = blockIdx.x;
  const float cx = centers[kc * 3], cy = centers[kc * 3 + 1], cz = centers[kc * 3 + 2];

  for (int i = tid; i < 192; i += 256) w1c[i] = W1[i];  // W1 rows 0..2 (rel weights)

  // Register-resident B fragments for W2 (8) and W3 (16), reused for all tiles.
  v16h w2f[2][4];
#pragma unroll
  for (int kt = 0; kt < 2; ++kt)
#pragma unroll
    for (int nt = 0; nt < 4; ++nt)
      w2f[kt][nt] = *(const v16h*)(pw2 + ((kt * 4 + nt) * 32 + lane) * 16);
  v16h w3f[2][8];
#pragma unroll
  for (int kt = 0; kt < 2; ++kt)
#pragma unroll
    for (int nt = 0; nt < 8; ++nt)
      w3f[kt][nt] = *(const v16h*)(pw3 + ((kt * 8 + nt) * 32 + lane) * 16);

  float bias2[4], bias3[8];
#pragma unroll
  for (int nt = 0; nt < 4; ++nt) bias2[nt] = b2v[nt * 16 + col];
#pragma unroll
  for (int nt = 0; nt < 8; ++nt) bias3[nt] = b3v[nt * 16 + col];

  v8f outAcc[8];
#pragma unroll
  for (int nt = 0; nt < 8; ++nt)
#pragma unroll
    for (int v = 0; v < 8; ++v) outAcc[nt][v] = NEG_INF;

  for (int p0 = 0; p0 < NPTS; p0 += 128) {
    __syncthreads();  // prev stage3 done with h2/maskAdd

    // ---- stage 1: h1 = relu(P + rel @ W1c) -> f16 LDS; ball mask ----------
    {
      const int lp = tid >> 1;            // local point 0..127
      const int p = p0 + lp;
      const int c0 = (tid & 1) * 32;      // 32 channels per thread
      float rx = coords[p * 3 + 0] - cx;
      float ry = coords[p * 3 + 1] - cy;
      float rz = coords[p * 3 + 2] - cz;
      if ((tid & 1) == 0)
        maskAdd[lp] = (rx * rx + ry * ry + rz * rz < 1.0f) ? 0.0f : NEG_INF;
      const float* Pp = P + (size_t)p * 64 + c0;
      if (p0 + 128 < NPTS)                // warm L2/L0 for the next tile
        __builtin_prefetch(P + (size_t)(p + 128) * 64 + c0, 0, 3);
#pragma unroll
      for (int c = 0; c < 32; ++c) {
        float v = Pp[c] + rx * w1c[c0 + c] + ry * w1c[64 + c0 + c] + rz * w1c[128 + c0 + c];
        h1ls[lp * H1STR + c0 + c] = (_Float16)fmaxf(v, 0.0f);
      }
    }
    __syncthreads();

    // ---- stage 2: h2 = relu(h1 @ W2 + b2) -> f16 LDS ------------------------
    {
      v8f acc[4];
#pragma unroll
      for (int nt = 0; nt < 4; ++nt)
#pragma unroll
        for (int v = 0; v < 8; ++v) acc[nt][v] = bias2[nt];

      const _Float16* rowp = h1ls + (wv * 16 + col) * H1STR;
#pragma unroll
      for (int kt = 0; kt < 2; ++kt) {
        int c0 = kt * 32 + 8 * half;
        v16h a = afrag_lds(rowp + c0, rowp + c0 + 16);
#pragma unroll
        for (int nt = 0; nt < 4; ++nt) acc[nt] = wmma16(a, w2f[kt][nt], acc[nt]);
      }
#pragma unroll
      for (int nt = 0; nt < 4; ++nt)
#pragma unroll
        for (int v = 0; v < 8; ++v)
          h2ls[(wv * 16 + v + 8 * half) * H1STR + nt * 16 + col] =
              (_Float16)fmaxf(acc[nt][v], 0.0f);
    }
    __syncthreads();

    // ---- stage 3: h3 = relu(h2 @ W3 + b3); masked running max (no store) ----
    {
      v8f acc[8];
#pragma unroll
      for (int nt = 0; nt < 8; ++nt)
#pragma unroll
        for (int v = 0; v < 8; ++v) acc[nt][v] = bias3[nt];

      const _Float16* rowp = h2ls + (wv * 16 + col) * H1STR;
#pragma unroll
      for (int kt = 0; kt < 2; ++kt) {
        int c0 = kt * 32 + 8 * half;
        v16h a = afrag_lds(rowp + c0, rowp + c0 + 16);
#pragma unroll
        for (int nt = 0; nt < 8; ++nt) acc[nt] = wmma16(a, w3f[kt][nt], acc[nt]);
      }
#pragma unroll
      for (int nt = 0; nt < 8; ++nt)
#pragma unroll
        for (int v = 0; v < 8; ++v) {
          float m = maskAdd[wv * 16 + v + 8 * half];     // 0 or -inf
          float val = fmaxf(acc[nt][v], 0.0f) + m;
          outAcc[nt][v] = fmaxf(outAcc[nt][v], val);
        }
    }
  }

  // ---- final reduce: rows within fragment, lane halves, then waves ----------
  __syncthreads();
#pragma unroll
  for (int nt = 0; nt < 8; ++nt) {
    float m = outAcc[nt][0];
#pragma unroll
    for (int v = 1; v < 8; ++v) m = fmaxf(m, outAcc[nt][v]);
    m = fmaxf(m, __shfl_xor(m, 16, 32));   // combine M and M+8 halves
    if (lane < 16) red[wv][nt * 16 + lane] = m;
  }
  __syncthreads();
  if (tid < L3DIM) {
    float r = red[0][tid];
#pragma unroll
    for (int w = 1; w < 8; ++w) r = fmaxf(r, red[w][tid]);
    outFeat[(size_t)kc * L3DIM + tid] = r;
  }
}

// ---------------------------------------------------------------------------
// Launch
// ---------------------------------------------------------------------------
extern "C" void kernel_launch(void* const* d_in, const int* in_sizes, int n_in,
                              void* d_out, int out_size, void* d_ws, size_t ws_size,
                              hipStream_t stream) {
  const float* coords   = (const float*)d_in[0];
  const float* features = (const float*)d_in[1];
  const float* W1 = (const float*)d_in[2];
  const float* b1 = (const float*)d_in[3];
  const float* W2 = (const float*)d_in[4];
  const float* b2 = (const float*)d_in[5];
  const float* W3 = (const float*)d_in[6];
  const float* b3 = (const float*)d_in[7];
  float* out = (float*)d_out;

  char* ws = (char*)d_ws;
  float*    centersWs = (float*)(ws + 0);           // 128*3 f32
  _Float16* pw1 = (_Float16*)(ws + 4096);           // 4096 f16
  _Float16* pw2 = (_Float16*)(ws + 4096 + 8192);    // 4096 f16
  _Float16* pw3 = (_Float16*)(ws + 4096 + 16384);   // 8192 f16
  float*    P   = (float*)(ws + 40960);             // 8192*64 f32 (2 MB)

  fps_kernel<<<1, 1024, 0, stream>>>(coords, centersWs, out);           // centers -> out[0:384]
  pack_weights<<<48, 256, 0, stream>>>(W1, W2, W3, pw1, pw2, pw3);
  mlp1_feat<<<64, 256, 0, stream>>>(features, b1, pw1, P);
  sa_fused<<<KCEN, 256, 0, stream>>>(coords, P, W1, b2, b3, pw2, pw3,
                                     centersWs, out + KCEN * 3);
}
